// ViS4mer_26268019982998
// MI455X (gfx1250) — matrix operand order repaired and checked by reading, not compiled, not run on hardware
//
#include <hip/hip_runtime.h>
#include <hip/hip_bf16.h>
#include <math.h>

// ---------------------------------------------------------------------------
// Types for WMMA fragments (gfx1250: v_wmma_f32_16x16x32_bf16)
// ---------------------------------------------------------------------------
typedef __attribute__((ext_vector_type(16))) __bf16          v16bf;
typedef __attribute__((ext_vector_type(8)))  float           v8f;
typedef __attribute__((ext_vector_type(8)))  unsigned short  v8us;
typedef __attribute__((ext_vector_type(4)))  unsigned short  v4us;

union FragBF { v8us h[2]; v16bf bf; };

__device__ __forceinline__ unsigned short f2bf(float f) {
    unsigned int u = __float_as_uint(f);
    unsigned int r = u + 0x7fffu + ((u >> 16) & 1u);   // round to nearest even
    return (unsigned short)(r >> 16);
}

__device__ __forceinline__ float gelu_f(float v) {
    return 0.5f * v * (1.0f + erff(v * 0.70710678118654752f));
}

// ---------------------------------------------------------------------------
// 0) Transpose input (B, L, D) f32 -> (B, D, L) f32
// ---------------------------------------------------------------------------
__global__ __launch_bounds__(256) void transpose_in_kernel(
    const float* __restrict__ x, float* __restrict__ xT, int L, int D)
{
    __shared__ float tile[32][33];
    const int b  = blockIdx.z;
    const int l0 = blockIdx.x * 32;
    const int d0 = blockIdx.y * 32;
    const int tx = threadIdx.x;           // 0..31
    const int ty = threadIdx.y;           // 0..7
    const float* xb = x  + (size_t)b * L * D;
    float*       xo = xT + (size_t)b * D * L;
#pragma unroll
    for (int i = 0; i < 32; i += 8)
        tile[ty + i][tx] = xb[(size_t)(l0 + ty + i) * D + d0 + tx];
    __syncthreads();
#pragma unroll
    for (int i = 0; i < 32; i += 8)
        xo[(size_t)(d0 + ty + i) * L + l0 + tx] = tile[tx][ty + i];
}

// ---------------------------------------------------------------------------
// 1) f32 -> bf16 convert (weights)
// ---------------------------------------------------------------------------
__global__ __launch_bounds__(256) void convert_bf16_kernel(
    const float* __restrict__ src, unsigned short* __restrict__ dst, int n)
{
    int i = blockIdx.x * 256 + threadIdx.x;
    if (i < n) dst[i] = f2bf(src[i]);
}

// ---------------------------------------------------------------------------
// 2) LayerNorm over channel dim H of (B, H, L). One block: 32 l-columns.
// ---------------------------------------------------------------------------
__global__ __launch_bounds__(256) void layernorm_kernel(
    const float* __restrict__ x, const float* __restrict__ g,
    const float* __restrict__ bb, float* __restrict__ z, int H, int L)
{
    const int tl = threadIdx.x & 31;
    const int th = threadIdx.x >> 5;      // 0..7
    const int b  = blockIdx.y;
    const int l  = blockIdx.x * 32 + tl;
    const float* xb = x + (size_t)b * H * L;
    float s = 0.f, q = 0.f;
    for (int h = th; h < H; h += 8) {
        float v = xb[(size_t)h * L + l];
        s += v; q = fmaf(v, v, q);
    }
    __shared__ float S[8][32], Q[8][32];
    S[th][tl] = s; Q[th][tl] = q;
    __syncthreads();
    if (th == 0) {
        float ss = 0.f, qq = 0.f;
#pragma unroll
        for (int r = 0; r < 8; ++r) { ss += S[r][tl]; qq += Q[r][tl]; }
        float mu  = ss / (float)H;
        float var = qq / (float)H - mu * mu;
        S[0][tl] = mu;
        Q[0][tl] = rsqrtf(var + 1e-5f);
    }
    __syncthreads();
    const float mu = S[0][tl], rstd = Q[0][tl];
    float* zb = z + (size_t)b * H * L;
    for (int h = th; h < H; h += 8) {
        float v = xb[(size_t)h * L + l];
        zb[(size_t)h * L + l] = (v - mu) * rstd * g[h] + bb[h];
    }
}

// ---------------------------------------------------------------------------
// 3) S4D causal conv via diagonal-SSM recurrence. One wave32 per (b,h).
//    Lane owns states n=lane and n=lane+32 (NSTATE=64).
//    Fully-unrolled 32-step chunks: z broadcast by v_readlane (immediate
//    lane), per-step n-reduction deferred into one 32x32 LDS transpose per
//    chunk (per-wave private region -> no barriers, same-wave DS is in-order).
//    Output GELU(conv + Dskip*z) written bf16 in (B,H,L).
// ---------------------------------------------------------------------------
__global__ __launch_bounds__(256) void s4d_scan_kernel(
    const float* __restrict__ z, const float* __restrict__ logdt,
    const float* __restrict__ Are, const float* __restrict__ Aim,
    const float* __restrict__ Cre, const float* __restrict__ Cim,
    const float* __restrict__ Dskip, unsigned short* __restrict__ ybf,
    int H, int L)
{
    const int lane = threadIdx.x & 31;
    const int wid  = threadIdx.x >> 5;
    const int gw   = blockIdx.x * 8 + wid;                 // global wave id
    const int b    = gw / H;
    const int h    = gw - b * H;

    __shared__ float red_all[8][32 * 36];                  // per-wave region
    float* red = red_all[wid];

    const float dt = expf(logdt[h]);
    float wr[2], wi[2], tr[2], ti[2];
#pragma unroll
    for (int j = 0; j < 2; ++j) {
        const int n = lane + 32 * j;
        float ar = Are[(size_t)h * 64 + n];
        float ai = Aim[(size_t)h * 64 + n];
        float dr = ar * dt, di = ai * dt;
        float e  = expf(dr);
        float cwr = e * cosf(di), cwi = e * sinf(di);     // w = exp(dt*A)
        wr[j] = cwr; wi[j] = cwi;
        float emr = cwr - 1.0f, emi = cwi;                // exp(dt*A) - 1
        float inv = 1.0f / (ar * ar + ai * ai);
        float qr  = (emr * ar + emi * ai) * inv;          // (e-1)/A
        float qi  = (emi * ar - emr * ai) * inv;
        float cr  = Cre[(size_t)h * 64 + n];
        float ci  = Cim[(size_t)h * 64 + n];
        tr[j] = 2.0f * (cr * qr - ci * qi);               // 2*Ct
        ti[j] = 2.0f * (cr * qi + ci * qr);
    }

    float sr0 = 0.f, si0 = 0.f, sr1 = 0.f, si1 = 0.f;
    const float Dk = Dskip[h];
    const float*       zrow = z   + (size_t)b * H * L + (size_t)h * L;
    unsigned short*    yrow = ybf + (size_t)b * H * L + (size_t)h * L;

    for (int l0 = 0; l0 < L; l0 += 32) {
        const float zc = zrow[l0 + lane];
        float p[32];
#pragma unroll
        for (int t = 0; t < 32; ++t) {
            float zv = __uint_as_float(
                __builtin_amdgcn_readlane(__float_as_uint(zc), t));
            float nsr0 = fmaf(wr[0], sr0, fmaf(-wi[0], si0, zv));
            float nsi0 = fmaf(wr[0], si0, wi[0] * sr0);
            float nsr1 = fmaf(wr[1], sr1, fmaf(-wi[1], si1, zv));
            float nsi1 = fmaf(wr[1], si1, wi[1] * sr1);
            sr0 = nsr0; si0 = nsi0; sr1 = nsr1; si1 = nsi1;
            p[t] = fmaf(tr[0], sr0, fmaf(-ti[0], si0,
                   fmaf(tr[1], sr1, -ti[1] * si1)));
        }
        // 32x32 transpose-reduce through LDS (per-wave private; in-order DS)
#pragma unroll
        for (int t = 0; t < 32; ++t) red[t * 36 + lane] = p[t];
        float acc = 0.f;
#pragma unroll
        for (int jj = 0; jj < 8; ++jj) {
            float4 v4 = *(const float4*)(red + lane * 36 + jj * 4);
            acc += (v4.x + v4.y) + (v4.z + v4.w);
        }
        float yv = acc + Dk * zc;
        yrow[l0 + lane] = f2bf(gelu_f(yv));
    }
}

// ---------------------------------------------------------------------------
// 4) GLU GEMM: a = Wglu[0:H]·y + bglu[0:H]; c = Wglu[H:2H]·y + bglu[H:2H]
//    v = a*sigmoid(c) + x   (residual fused). WMMA bf16, f32 accumulate.
//    Block tile 64(M a-rows + matching c-rows) x 64(N=l), K step 32.
//    8 waves: wm in 0..3 (16 rows each), wn in 0..1 (32 cols each).
// ---------------------------------------------------------------------------
__global__ __launch_bounds__(256) void glu_gemm_kernel(
    const unsigned short* __restrict__ Wbf, const float* __restrict__ bglu,
    const unsigned short* __restrict__ ybf, const float* __restrict__ xin,
    float* __restrict__ vout, int H, int L)
{
    const int tid  = threadIdx.x;
    const int lane = tid & 31;
    const int wid  = tid >> 5;
    const int wm   = wid & 3;
    const int wn   = wid >> 2;
    const int b    = blockIdx.z;
    const int m0   = blockIdx.y * 64;
    const int n0   = blockIdx.x * 64;
    const int kh   = lane >> 4;
    const int l15  = lane & 15;

    __shared__ __align__(16) unsigned short Bs[64 * 40];  // [n][k], k-stride 40

    v8f acc_a0 = {0,0,0,0,0,0,0,0}, acc_a1 = {0,0,0,0,0,0,0,0};
    v8f acc_c0 = {0,0,0,0,0,0,0,0}, acc_c1 = {0,0,0,0,0,0,0,0};

    // ---- strength-reduced pointers (advance by +32 along K each step) ----
    const int kk2 = tid >> 4;             // 0..15 -> k rows {2*kk2, 2*kk2+1}
    const int nch = (tid & 15) * 4;       // 0..60
    const unsigned short* src0 =
        ybf + (size_t)b * H * L + (size_t)(2 * kk2) * L + n0 + nch;
    const unsigned short* src1 = src0 + L;
    unsigned int* Bs32 = (unsigned int*)Bs;

    const unsigned short* arow = Wbf + (size_t)(m0 + wm * 16 + l15) * H + 8 * kh;
    const unsigned short* crow = arow + (size_t)H * H;

    const unsigned short* pB0 = Bs + (wn * 32 + l15) * 40 + 8 * kh;
    const unsigned short* pB1 = pB0 + 16 * 40;
    const size_t kstepB = (size_t)32 * L;

    for (int k0 = 0; k0 < H; k0 += 32) {
        __syncthreads();
        {   // stage B tile [32 k][64 n] transposed into LDS [n][k], packed b32
            v4us r0 = *(const v4us*)src0;
            v4us r1 = *(const v4us*)src1;
#pragma unroll
            for (int e = 0; e < 4; ++e)
                Bs32[(nch + e) * 20 + kk2] =
                    (unsigned int)r0[e] | ((unsigned int)r1[e] << 16);
            src0 += kstepB; src1 += kstepB;
        }
        __syncthreads();

        FragBF fa, fc;                      // A-fragments: lane = M row
        fa.h[0] = *(const v8us*)(arow);
        fa.h[1] = *(const v8us*)(arow + 16);
        fc.h[0] = *(const v8us*)(crow);
        fc.h[1] = *(const v8us*)(crow + 16);
        arow += 32; crow += 32;

        FragBF fb0, fb1;                    // B-fragments: lane = N col
        fb0.h[0] = *(const v8us*)(pB0);
        fb0.h[1] = *(const v8us*)(pB0 + 16);
        fb1.h[0] = *(const v8us*)(pB1);
        fb1.h[1] = *(const v8us*)(pB1 + 16);

        acc_a0 = __builtin_amdgcn_wmma_f32_16x16x32_bf16(false, fa.bf, false, fb0.bf, (short)0, acc_a0, false, false);
        acc_a1 = __builtin_amdgcn_wmma_f32_16x16x32_bf16(false, fa.bf, false, fb1.bf, (short)0, acc_a1, false, false);
        acc_c0 = __builtin_amdgcn_wmma_f32_16x16x32_bf16(false, fc.bf, false, fb0.bf, (short)0, acc_c0, false, false);
        acc_c1 = __builtin_amdgcn_wmma_f32_16x16x32_bf16(false, fc.bf, false, fb1.bf, (short)0, acc_c1, false, false);
    }

    const float* xb = xin  + (size_t)b * H * L;
    float*       vb = vout + (size_t)b * H * L;
#pragma unroll
    for (int r = 0; r < 8; ++r) {
        const int row = m0 + wm * 16 + r + 8 * kh;   // C/D layout: M=r+8*(lane>=16)
        const float ba = bglu[row], bc = bglu[H + row];
        const int nA = n0 + wn * 32 + l15;
        const int nB = nA + 16;
        const size_t base = (size_t)row * L;
        float a0 = acc_a0[r] + ba, c0 = acc_c0[r] + bc;
        float a1 = acc_a1[r] + ba, c1 = acc_c1[r] + bc;
        vb[base + nA] = a0 * (1.f / (1.f + expf(-c0))) + xb[base + nA];
        vb[base + nB] = a1 * (1.f / (1.f + expf(-c1))) + xb[base + nB];
    }
}

// ---------------------------------------------------------------------------
// 5) AvgPool(2) over length + bf16 convert: (B,H,L) f32 -> (B,H,L/2) bf16
// ---------------------------------------------------------------------------
__global__ __launch_bounds__(256) void pool_kernel(
    const float* __restrict__ v, unsigned short* __restrict__ p,
    int total, int halfL)
{
    int i = blockIdx.x * 256 + threadIdx.x;
    if (i >= total) return;
    int row = i / halfL;
    int j   = i - row * halfL;
    float2 vv = ((const float2*)v)[(size_t)row * halfL + j];
    p[i] = f2bf(0.5f * (vv.x + vv.y));
}

// ---------------------------------------------------------------------------
// 6) Wlin GEMM: out = GELU(Wlin (M x K) · p (K x N) + blin), written f32.
// ---------------------------------------------------------------------------
__global__ __launch_bounds__(256) void lin_gemm_kernel(
    const unsigned short* __restrict__ Wbf, const float* __restrict__ bias,
    const unsigned short* __restrict__ pbf, float* __restrict__ outp,
    int M, int K, int N)
{
    const int tid  = threadIdx.x;
    const int lane = tid & 31;
    const int wid  = tid >> 5;
    const int wm   = wid & 3;
    const int wn   = wid >> 2;
    const int b    = blockIdx.z;
    const int m0   = blockIdx.y * 64;
    const int n0   = blockIdx.x * 64;
    const int kh   = lane >> 4;
    const int l15  = lane & 15;

    __shared__ __align__(16) unsigned short Bs[64 * 40];

    v8f acc0 = {0,0,0,0,0,0,0,0}, acc1 = {0,0,0,0,0,0,0,0};

    const int kk2 = tid >> 4;
    const int nch = (tid & 15) * 4;
    const unsigned short* src0 =
        pbf + (size_t)b * K * N + (size_t)(2 * kk2) * N + n0 + nch;
    const unsigned short* src1 = src0 + N;
    unsigned int* Bs32 = (unsigned int*)Bs;

    const unsigned short* arow = Wbf + (size_t)(m0 + wm * 16 + l15) * K + 8 * kh;

    const unsigned short* pB0 = Bs + (wn * 32 + l15) * 40 + 8 * kh;
    const unsigned short* pB1 = pB0 + 16 * 40;
    const size_t kstepB = (size_t)32 * N;

    for (int k0 = 0; k0 < K; k0 += 32) {
        __syncthreads();
        {
            v4us r0 = *(const v4us*)src0;
            v4us r1 = *(const v4us*)src1;
#pragma unroll
            for (int e = 0; e < 4; ++e)
                Bs32[(nch + e) * 20 + kk2] =
                    (unsigned int)r0[e] | ((unsigned int)r1[e] << 16);
            src0 += kstepB; src1 += kstepB;
        }
        __syncthreads();

        FragBF fa;
        fa.h[0] = *(const v8us*)(arow);
        fa.h[1] = *(const v8us*)(arow + 16);
        arow += 32;

        FragBF fb0, fb1;
        fb0.h[0] = *(const v8us*)(pB0);
        fb0.h[1] = *(const v8us*)(pB0 + 16);
        fb1.h[0] = *(const v8us*)(pB1);
        fb1.h[1] = *(const v8us*)(pB1 + 16);

        acc0 = __builtin_amdgcn_wmma_f32_16x16x32_bf16(false, fa.bf, false, fb0.bf, (short)0, acc0, false, false);
        acc1 = __builtin_amdgcn_wmma_f32_16x16x32_bf16(false, fa.bf, false, fb1.bf, (short)0, acc1, false, false);
    }

    float* ob = outp + (size_t)b * M * N;
#pragma unroll
    for (int r = 0; r < 8; ++r) {
        const int row = m0 + wm * 16 + r + 8 * kh;
        const float bs = bias[row];
        const int nA = n0 + wn * 32 + l15;
        const size_t base = (size_t)row * N;
        ob[base + nA]      = gelu_f(acc0[r] + bs);
        ob[base + nA + 16] = gelu_f(acc1[r] + bs);
    }
}

// ---------------------------------------------------------------------------
// 7) Decoder: mean over L, then (10 x 128) linear. One block per batch.
// ---------------------------------------------------------------------------
__global__ __launch_bounds__(128) void decoder_kernel(
    const float* __restrict__ x, const float* __restrict__ Wdec,
    const float* __restrict__ bdec, float* __restrict__ out, int H, int L)
{
    __shared__ float xm[128];
    const int b = blockIdx.x;
    const int h = threadIdx.x;
    const float* row = x + (size_t)b * H * L + (size_t)h * L;
    float s = 0.f;
    for (int l = 0; l < L; ++l) s += row[l];
    xm[h] = s / (float)L;
    __syncthreads();
    if (h < 10) {
        float acc = bdec[h];
        for (int k = 0; k < H; ++k) acc = fmaf(Wdec[h * H + k], xm[k], acc);
        out[b * 10 + h] = acc;
    }
}

// ---------------------------------------------------------------------------
// Host-side launch
// ---------------------------------------------------------------------------
extern "C" void kernel_launch(void* const* d_in, const int* in_sizes, int n_in,
                              void* d_out, int out_size, void* d_ws, size_t ws_size,
                              hipStream_t stream) {
    (void)in_sizes; (void)n_in; (void)out_size; (void)ws_size;
    constexpr int    BATCH = 16, L0 = 2048, D0 = 1024, NL = 3;
    constexpr size_t SZ_A = (size_t)BATCH * D0 * L0 * 4;          // 134 MB
    constexpr size_t SZ_B = (size_t)BATCH * (D0/2) * (L0/2) * 4;  // 33.5 MB
    constexpr size_t SZ_Z = SZ_A;                                 // z / v
    constexpr size_t SZ_Y = (size_t)BATCH * D0 * L0 * 2;          // bf16 y / p

    char* ws = (char*)d_ws;
    float*          bufA = (float*)(ws);
    float*          bufB = (float*)(ws + SZ_A);
    float*          bufZ = (float*)(ws + SZ_A + SZ_B);
    unsigned short* bufY = (unsigned short*)(ws + SZ_A + SZ_B + SZ_Z);
    unsigned short* bufW = (unsigned short*)(ws + SZ_A + SZ_B + SZ_Z + SZ_Y);

    const float* x_in = (const float*)d_in[0];

    // (B, L, D) -> (B, D, L)
    transpose_in_kernel<<<dim3(L0/32, D0/32, BATCH), dim3(32, 8), 0, stream>>>(
        x_in, bufA, L0, D0);

    float* xcur = bufA;
    float* xnxt = bufB;
    int H = D0, L = L0;
    for (int l = 0; l < NL; ++l) {
        const int base = 1 + 12 * l;
        const float* g     = (const float*)d_in[base + 0];
        const float* bvec  = (const float*)d_in[base + 1];
        const float* logdt = (const float*)d_in[base + 2];
        const float* Are   = (const float*)d_in[base + 3];
        const float* Aim   = (const float*)d_in[base + 4];
        const float* Cre   = (const float*)d_in[base + 5];
        const float* Cim   = (const float*)d_in[base + 6];
        const float* Dskip = (const float*)d_in[base + 7];
        const float* Wglu  = (const float*)d_in[base + 8];
        const float* bglu  = (const float*)d_in[base + 9];
        const float* Wlin  = (const float*)d_in[base + 10];
        const float* blin  = (const float*)d_in[base + 11];

        unsigned short* WgluBF = bufW;
        unsigned short* WlinBF = bufW + (size_t)2 * H * H;
        const int nglu = 2 * H * H;
        const int nlin = (H / 2) * H;
        convert_bf16_kernel<<<(nglu + 255) / 256, 256, 0, stream>>>(Wglu, WgluBF, nglu);
        convert_bf16_kernel<<<(nlin + 255) / 256, 256, 0, stream>>>(Wlin, WlinBF, nlin);

        layernorm_kernel<<<dim3(L / 32, BATCH), 256, 0, stream>>>(
            xcur, g, bvec, bufZ, H, L);

        s4d_scan_kernel<<<(BATCH * H) / 8, 256, 0, stream>>>(
            bufZ, logdt, Are, Aim, Cre, Cim, Dskip, bufY, H, L);

        glu_gemm_kernel<<<dim3(L / 64, H / 64, BATCH), 256, 0, stream>>>(
            WgluBF, bglu, bufY, xcur, bufZ, H, L);

        const int halfL = L / 2;
        const int total = BATCH * H * halfL;
        pool_kernel<<<(total + 255) / 256, 256, 0, stream>>>(bufZ, bufY, total, halfL);

        lin_gemm_kernel<<<dim3(halfL / 64, (H / 2) / 64, BATCH), 256, 0, stream>>>(
            WlinBF, blin, bufY, xnxt, H / 2, H, halfL);

        float* t = xcur; xcur = xnxt; xnxt = t;
        H >>= 1; L >>= 1;
    }

    const float* Wdec = (const float*)d_in[1 + 12 * NL];
    const float* bdec = (const float*)d_in[2 + 12 * NL];
    decoder_kernel<<<BATCH, 128, 0, stream>>>(xcur, Wdec, bdec, (float*)d_out, H, L);
}
